// QuantLinear_33492154974427
// MI455X (gfx1250) — matrix operand order repaired and checked by reading, not compiled, not run on hardware
//
#include <hip/hip_runtime.h>

// ---------------------------------------------------------------------------
// GPTQ W4A16 dequant + GEMM for gfx1250 (MI455X), wave32 + WMMA.
//   A:       (16, 4096)  fp16
//   qweight: (11008, 2048) int32, each holding one packed byte (2 nibbles)
//   scales:  (11008, 32) fp16   zeros: (11008, 32) fp16   bias: (11008,) fp16
//   out = A @ dequant(qweight).T + bias    -> (16, 11008) fp16
// HBM-bound (~90MB qweight); one 16x16 tile per block, 4 waves K-split.
// ---------------------------------------------------------------------------

typedef _Float16 half_t;
typedef __attribute__((ext_vector_type(2)))  _Float16 h2;
typedef __attribute__((ext_vector_type(8)))  _Float16 v8h;
typedef __attribute__((ext_vector_type(16))) _Float16 v16h;
typedef __attribute__((ext_vector_type(8)))  float    v8f;

#define Mdim   16
#define Kdim   4096
#define Ndim   11008
#define GRP    128
#define KSPLIT 4              // waves per block, each owns K/KSPLIT
#define KW     (Kdim / KSPLIT)

union H2U { unsigned int u; h2 h; };
union BU  { v16h v; unsigned int w[8]; };
union AU  { v16h v; v8h h[2]; };

__global__ __launch_bounds__(128) void gptq_wmma_kernel(
    const half_t* __restrict__ A,
    const unsigned int* __restrict__ qw,
    const half_t* __restrict__ scales,
    const half_t* __restrict__ zeros,
    const half_t* __restrict__ bias,
    half_t* __restrict__ out)
{
    __shared__ float red[KSPLIT][Mdim][16];

    const int tid  = threadIdx.x;
    const int lane = tid & 31;
    const int wave = tid >> 5;
    const int nloc = lane & 15;   // N column within tile (B/C lane layout)
    const int hid  = lane >> 4;   // which K half-group this lane holds

    const int tile = blockIdx.x;
    const int ncol = tile * 16 + nloc;

    const unsigned int* qrow = qw + (size_t)ncol * (Kdim / 2);  // 2048 dwords/row
    const half_t*       arow = A  + (size_t)nloc * Kdim;        // M row = lane%16

    const int kw0 = wave * KW;

    v8f acc = {};

    for (int grp = 0; grp < KW / GRP; ++grp) {
        const int kg   = kw0 + grp * GRP;
        const int gidx = kg >> 7;  // global quant-group index
        const half_t s = scales[ncol * (Kdim / GRP) + gidx];
        const half_t z = zeros [ncol * (Kdim / GRP) + gidx];
        h2 s2;  s2[0]  = s;  s2[1]  = s;
        // magic-bias dequant: half(0x6400|q) == 1024+q, so subtract (1024+z)
        const half_t zb = (half_t)((float)z + 1024.0f);
        h2 zb2; zb2[0] = zb; zb2[1] = zb;

        if (grp + 1 < KW / GRP) {
            __builtin_prefetch(qrow + ((kg + GRP) >> 1), 0, 1); // global_prefetch
        }

        #pragma unroll
        for (int it = 0; it < GRP / 32; ++it) {
            const int k0 = kg + it * 32;

            // ---- A tile 16x32 f16 in WMMA-A layout ----
            // lanes 0-15: K = k0+[0..15]; lanes 16-31: k0+16+[0..15]
            AU a;
            a.h[0] = *(const v8h*)(arow + k0 +     hid * 16);
            a.h[1] = *(const v8h*)(arow + k0 + 8 + hid * 16);

            // ---- B tile 32x16: dequant 16 nibbles this lane owns ----
            // lane col = nloc, K base = k0 + hid*16, 8 dwords = 8 bytes packed
            const int kb = k0 + hid * 16;
            const unsigned int* qp = qrow + (kb >> 1);
            uint4 q0 = *(const uint4*)(qp);
            uint4 q1 = *(const uint4*)(qp + 4);
            unsigned int qq[8] = {q0.x, q0.y, q0.z, q0.w, q1.x, q1.y, q1.z, q1.w};

            BU b;
            #pragma unroll
            for (int j = 0; j < 8; ++j) {
                unsigned int t = (qq[j] & 0xFu) | ((qq[j] & 0xF0u) << 12);
                t |= 0x64006400u;                 // -> half2 {1024+lo, 1024+hi}
                H2U c; c.u = t;
                h2 w = (c.h - zb2) * s2;          // v_pk_add_f16 + v_pk_mul_f16
                H2U o; o.h = w;
                b.w[j] = o.u;
            }

            acc = __builtin_amdgcn_wmma_f32_16x16x32_f16(
                false, a.v, false, b.v, (short)0, acc, false, false);
        }
    }

    // ---- cross-wave K reduction in LDS ----
    // C layout: lane l, vgpr v -> M = v + (l/16)*8, N = l%16
    #pragma unroll
    for (int v = 0; v < 8; ++v) {
        red[wave][v + hid * 8][nloc] = acc[v];
    }
    __syncthreads();

    for (int e = tid; e < Mdim * 16; e += 128) {
        const int m  = e >> 4;
        const int nn = e & 15;
        float sum = red[0][m][nn] + red[1][m][nn] + red[2][m][nn] + red[3][m][nn];
        const int ng = tile * 16 + nn;
        sum += (float)bias[ng];
        out[(size_t)m * Ndim + ng] = (half_t)sum;
    }
}

extern "C" void kernel_launch(void* const* d_in, const int* in_sizes, int n_in,
                              void* d_out, int out_size, void* d_ws, size_t ws_size,
                              hipStream_t stream)
{
    (void)in_sizes; (void)n_in; (void)out_size; (void)d_ws; (void)ws_size;

    const half_t*       A  = (const half_t*)d_in[0];
    const unsigned int* q  = (const unsigned int*)d_in[1];
    const half_t*       sc = (const half_t*)d_in[2];
    const half_t*       zp = (const half_t*)d_in[3];
    const half_t*       bi = (const half_t*)d_in[4];
    half_t*             o  = (half_t*)d_out;

    dim3 grid(Ndim / 16);   // 688 tiles
    dim3 block(128);        // 4 wave32 = K-split of 4
    hipLaunchKernelGGL(gptq_wmma_kernel, grid, block, 0, stream,
                       A, q, sc, zp, bi, o);
}
